// ContinuousFilterConvolution_24592982736978
// MI455X (gfx1250) — matrix operand order repaired and checked by reading, not compiled, not run on hardware
//
#include <hip/hip_runtime.h>
#include <hip/hip_bf16.h>

// CDNA5 / gfx1250, wave32. WMMA 16x16x32 bf16, f32 accumulation.
// One wave processes 32 edges (two 16-row M-tiles) so every B fragment
// loaded from L2 feeds two v_wmma instructions.

typedef __attribute__((ext_vector_type(16))) __bf16 v16bf;
typedef __attribute__((ext_vector_type(8)))  __bf16 v8bf;
typedef __attribute__((ext_vector_type(8)))  float  v8f;

#define HIDDEN 256
#define NBASES 64
#define WPB 2          // waves per block (64 threads)

// Single-instruction relu: v_med3_f32(x, 0, FLT_MAX)
__device__ __forceinline__ float relu1(float x) {
    return __builtin_amdgcn_fmed3f(x, 0.0f, 3.402823466e+38f);
}

// ---------------------------------------------------------------------------
__global__ void zero_out_kernel(float* __restrict__ H, int n) {
    int i = blockIdx.x * blockDim.x + threadIdx.x;
    if (i < n) H[i] = 0.0f;
}

// ---------------------------------------------------------------------------
// Pack W1 (HIDDEN x NBASES, row-major f32) into bf16 WMMA B-layout.
// GEMM1: h[e,n] = sum_b rbf[e,b] * W1[n,b]  =>  B[k=b][n] = W1[n*64 + b]
// B-layout (16-bit, 32x16): lane l: n = l&15, half = l>>4; element e in [0,16):
//   k = (e>>3)*16 + half*8 + (e&7).   Buffer index: ((nt*2 + ks)*32 + lane)*16 + e.
__global__ void pack_w1_kernel(const float* __restrict__ W1, __bf16* __restrict__ B1p) {
    int i = blockIdx.x * blockDim.x + threadIdx.x;
    if (i >= 16 * 2 * 32 * 16) return;
    int e    = i & 15;
    int lane = (i >> 4) & 31;
    int ks   = (i >> 9) & 1;
    int nt   = i >> 10;
    int n = nt * 16 + (lane & 15);
    int b = ks * 32 + (e >> 3) * 16 + (lane >> 4) * 8 + (e & 7);
    B1p[i] = (__bf16)W1[n * NBASES + b];
}

// Pack W2 (HIDDEN x HIDDEN, row-major f32) into bf16 WMMA B-layout.
// GEMM2: M[e,g] = sum_h hr[e,h] * W2[g,h]  =>  B[k=h][n=g] = W2[g*256 + h]
__global__ void pack_w2_kernel(const float* __restrict__ W2, __bf16* __restrict__ B2p) {
    int i = blockIdx.x * blockDim.x + threadIdx.x;
    if (i >= 16 * 8 * 32 * 16) return;
    int e    = i & 15;
    int lane = (i >> 4) & 31;
    int ks   = (i >> 9) & 7;
    int nt   = i >> 12;
    int n = nt * 16 + (lane & 15);
    int k = ks * 32 + (e >> 3) * 16 + (lane >> 4) * 8 + (e & 7);
    B2p[i] = (__bf16)W2[n * HIDDEN + k];
}

// ---------------------------------------------------------------------------
// Main kernel: one wave owns 32 edges = two 16-row WMMA M-tiles.
//   A-layout (16-bit 16x32): lane l: m = l&15, half = l>>4; element e:
//     k = (e>>3)*16 + half*8 + (e&7)
//   C/D layout (f32 16x16): VGPR r, lane l: n = l&15, m = r + 8*(l>>4)
__global__ __launch_bounds__(64)
void cfconv_kernel(const float* __restrict__ nf,
                   const float* __restrict__ coords,
                   const int*   __restrict__ src,
                   const int*   __restrict__ dst,
                   const __bf16* __restrict__ B1p,
                   const __bf16* __restrict__ B2p,
                   float* __restrict__ H,
                   int E, int T2) {
    // per-wave h staging: 32 rows x 256 cols bf16 (tile A rows 0-15, tile B 16-31)
    __shared__ __align__(16) __bf16 h_lds[WPB][32 * HIDDEN];
    __shared__ float d_sh[WPB][32];
    __shared__ int   s_sh[WPB][32];
    __shared__ int   t_sh[WPB][32];

    const int lane = threadIdx.x & 31;
    const int w    = threadIdx.x >> 5;
    const int half = lane >> 4;
    const int nl   = lane & 15;

    const int gw      = blockIdx.x * WPB + w;
    const int gstride = gridDim.x * WPB;

    const float inv63 = 1.0f / 63.0f;     // RBF center spacing
    const float gamma = 3969.0f;          // 63^2

    for (int t = gw; t < T2; t += gstride) {
        const int e0 = t * 32;

        // ---- per-edge distance + indices (all 32 lanes) ----
        {
            int e = e0 + lane;
            // Invalid (padding) edges get a huge distance: -gamma*u^2 -> -inf,
            // v_exp_f32 -> 0.0, so their A rows are exactly zero.
            float d = 1e19f; int s = 0, dd = 0;
            if (e < E) {
                s  = src[e];
                dd = dst[e];
                float dx = coords[s * 3 + 0] - coords[dd * 3 + 0];
                float dy = coords[s * 3 + 1] - coords[dd * 3 + 1];
                float dz = coords[s * 3 + 2] - coords[dd * 3 + 2];
                d = sqrtf(dx * dx + dy * dy + dz * dz);
            }
            d_sh[w][lane] = d;
            s_sh[w][lane] = s;
            t_sh[w][lane] = dd;
        }
        __builtin_amdgcn_wave_barrier();
        asm volatile("s_wait_dscnt 0" ::: "memory");

        const float dA = d_sh[w][nl];
        const float dB = d_sh[w][16 + nl];

        // ---- RBF A fragments, built directly in WMMA A-layout ----
        v16bf a0A, a1A, a0B, a1B;
#pragma unroll
        for (int e2 = 0; e2 < 16; ++e2) {
            int kk = (e2 >> 3) * 16 + half * 8 + (e2 & 7);   // k within K-step
            float c0 = (float)kk * inv63;
            float c1 = (float)(kk + 32) * inv63;
            float uA0 = dA - c0, uA1 = dA - c1;
            float uB0 = dB - c0, uB1 = dB - c1;
            a0A[e2] = (__bf16)__expf(-gamma * uA0 * uA0);
            a1A[e2] = (__bf16)__expf(-gamma * uA1 * uA1);
            a0B[e2] = (__bf16)__expf(-gamma * uB0 * uB0);
            a1B[e2] = (__bf16)__expf(-gamma * uB1 * uB1);
        }

        // ---- GEMM1: h = relu(rbf @ W1^T); each B fragment feeds 2 WMMAs ----
#pragma unroll
        for (int nt = 0; nt < 16; ++nt) {
            const v16bf b0 = *(const v16bf*)(B1p + ((size_t)((nt * 2 + 0) * 32 + lane)) * 16);
            const v16bf b1 = *(const v16bf*)(B1p + ((size_t)((nt * 2 + 1) * 32 + lane)) * 16);
            v8f accA = {}, accB = {};
            accA = __builtin_amdgcn_wmma_f32_16x16x32_bf16(false, a0A, false, b0, (short)0, accA, false, false);
            accB = __builtin_amdgcn_wmma_f32_16x16x32_bf16(false, a0B, false, b0, (short)0, accB, false, false);
            accA = __builtin_amdgcn_wmma_f32_16x16x32_bf16(false, a1A, false, b1, (short)0, accA, false, false);
            accB = __builtin_amdgcn_wmma_f32_16x16x32_bf16(false, a1B, false, b1, (short)0, accB, false, false);
#pragma unroll
            for (int r = 0; r < 8; ++r) {
                float vA = relu1(accA[r]);                        // single v_med3_f32
                float vB = relu1(accB[r]);
                h_lds[w][(r + 8 * half) * HIDDEN + nt * 16 + nl]      = (__bf16)vA;
                h_lds[w][(16 + r + 8 * half) * HIDDEN + nt * 16 + nl] = (__bf16)vB;
            }
        }
        __builtin_amdgcn_wave_barrier();
        asm volatile("s_wait_dscnt 0" ::: "memory");

        // ---- reload h tiles from LDS in A-layout (8 K-steps x 2 M-tiles) ----
        v16bf A2[2][8];
#pragma unroll
        for (int ks = 0; ks < 8; ++ks) {
#pragma unroll
            for (int mt = 0; mt < 2; ++mt) {
                const __bf16* base = &h_lds[w][(mt * 16 + nl) * HIDDEN + ks * 32 + half * 8];
                v8bf lo = *(const v8bf*)(base);          // k = ks*32 + half*8 + [0,8)
                v8bf hi = *(const v8bf*)(base + 16);     // k = ks*32 + 16 + half*8 + [0,8)
                A2[mt][ks] = __builtin_shufflevector(lo, hi, 0, 1, 2, 3, 4, 5, 6, 7,
                                                             8, 9, 10, 11, 12, 13, 14, 15);
            }
        }

        // ---- GEMM2: M = relu(h @ W2^T); fused gather * nf + atomic scatter ----
        for (int nt = 0; nt < 16; ++nt) {
            v8f accA = {}, accB = {};
#pragma unroll
            for (int ks = 0; ks < 8; ++ks) {
                const v16bf b = *(const v16bf*)(B2p + ((size_t)((nt * 8 + ks) * 32 + lane)) * 16);
                accA = __builtin_amdgcn_wmma_f32_16x16x32_bf16(false, A2[0][ks], false, b, (short)0, accA, false, false);
                accB = __builtin_amdgcn_wmma_f32_16x16x32_bf16(false, A2[1][ks], false, b, (short)0, accB, false, false);
            }
            const int n = nt * 16 + nl;
#pragma unroll
            for (int r = 0; r < 8; ++r) {
                int m = r + 8 * half;
                if (e0 + m < E) {
                    float v = relu1(accA[r]);
                    int s  = s_sh[w][m];
                    int dd = t_sh[w][m];
                    atomicAdd(&H[(size_t)dd * HIDDEN + n], v * nf[(size_t)s * HIDDEN + n]);
                }
                if (e0 + 16 + m < E) {
                    float v = relu1(accB[r]);
                    int s  = s_sh[w][16 + m];
                    int dd = t_sh[w][16 + m];
                    atomicAdd(&H[(size_t)dd * HIDDEN + n], v * nf[(size_t)s * HIDDEN + n]);
                }
            }
        }
    }
}

// ---------------------------------------------------------------------------
extern "C" void kernel_launch(void* const* d_in, const int* in_sizes, int n_in,
                              void* d_out, int out_size, void* d_ws, size_t ws_size,
                              hipStream_t stream) {
    const float* nf     = (const float*)d_in[0];   // node_feats (V,256)
    const float* coords = (const float*)d_in[1];   // (V,3)
    const float* W1     = (const float*)d_in[2];   // (256,64)
    const float* W2     = (const float*)d_in[3];   // (256,256)
    const int*   src    = (const int*)d_in[5];
    const int*   dst    = (const int*)d_in[6];
    float*       H      = (float*)d_out;

    const int E  = in_sizes[5];
    const int T2 = (E + 31) / 32;     // 32-edge tiles

    __bf16* B1p = (__bf16*)d_ws;                       // 32 KB
    __bf16* B2p = (__bf16*)((char*)d_ws + 32768);      // 128 KB

    zero_out_kernel<<<(out_size + 255) / 256, 256, 0, stream>>>(H, out_size);
    pack_w1_kernel<<<64, 256, 0, stream>>>(W1, B1p);
    pack_w2_kernel<<<256, 256, 0, stream>>>(W2, B2p);

    int waves = T2 < 8192 ? T2 : 8192;
    if (waves < 1) waves = 1;
    int blocks = (waves + WPB - 1) / WPB;
    cfconv_kernel<<<blocks, 64, 0, stream>>>(nf, coords, src, dst, B1p, B2p, H, E, T2);
}
